// OutputPPBlock_10883447128787
// MI455X (gfx1250) — matrix operand order repaired and checked by reading, not compiled, not run on hardware
//
#include <hip/hip_runtime.h>
#include <hip/hip_bf16.h>

// ---------------------------------------------------------------------------
// Problem constants (from reference)
// ---------------------------------------------------------------------------
#define E_EDGES   800000
#define N_NODES   25000
#define N_PAD     25024      // round up to multiple of 32 (tile rows)
#define HIDDEN    256
#define NUM_RAD   6
#define LDS_STRIDE 264       // 256 + 8 bf16 pad -> 528B row stride, 16B aligned,
                             // rows land on distinct bank groups for b128 reads
#define LDS_HALF  (32 * LDS_STRIDE)

typedef __attribute__((ext_vector_type(16))) __bf16 v16bf;
typedef __attribute__((ext_vector_type(8)))  float  v8f;
typedef __attribute__((ext_vector_type(4)))  unsigned int v4u;

union FragBF { v16bf v; v4u u[2]; };

// ---------------------------------------------------------------------------
// Kernel 0: zero the node accumulator (must run every launch; harness does not
// re-poison workspace between replays and edge kernel accumulates into it).
// ---------------------------------------------------------------------------
__global__ void zero_kernel(float4* __restrict__ p, int n4) {
    int i = blockIdx.x * blockDim.x + threadIdx.x;
    if (i < n4) p[i] = make_float4(0.f, 0.f, 0.f, 0.f);
}

// ---------------------------------------------------------------------------
// Kernel 1: convert W_up (layer 0) and Ws[0..2] (layers 1..3) from f32 to bf16,
// pre-swizzled into WMMA B-fragment order so each lane's fragment is one
// contiguous 32-byte read.
//
// B-fragment order used here (contiguous K, same as A): lane l<16 = column n,
// holds K=0..15 (two per dword); lane l>=16 = column n, holds K=16..31.
// Fragment storage: Bpack[layer][kk][nb][lane][pos], pos = K offset 0..15.
// Math is D = A(nodes x K) * B(K x out), so B[k][n] = W[n][k] (W stored [out][in]).
// ---------------------------------------------------------------------------
__global__ void pack_weights_kernel(const float* __restrict__ W_up,
                                    const float* __restrict__ Ws,
                                    __bf16* __restrict__ Bpack) {
    int id = blockIdx.x * blockDim.x + threadIdx.x;
    if (id >= 4 * 65536) return;
    int pos   = id & 15;
    int lane  = (id >> 4)  & 31;
    int nb    = (id >> 9)  & 15;
    int kk    = (id >> 13) & 7;
    int layer = id >> 16;
    int k = kk * 32 + ((lane < 16) ? 0 : 16) + pos;
    int n = nb * 16 + (lane & 15);
    float v = (layer == 0) ? W_up[n * HIDDEN + k]
                           : Ws[((layer - 1) * HIDDEN + n) * HIDDEN + k];
    Bpack[id] = (__bf16)v;
}

// ---------------------------------------------------------------------------
// Kernel 2: edge phase.  h_nodes[i[e]][c] += (sum_r rbf[e][r]*W_rbf[c][r]) * x[e][c]
// Memory bound: 819 MB of x reads dominates (~35us at 23.3 TB/s).  64 threads
// per edge, float4 per thread, W_rbf staged in LDS, f32 atomics into the
// L2-resident 25.6MB accumulator (global_atomic_add_f32 via unsafeAtomicAdd).
// ---------------------------------------------------------------------------
__global__ __launch_bounds__(256)
void edge_kernel(const float* __restrict__ x,
                 const float* __restrict__ rbf,
                 const int*   __restrict__ idx,
                 const float* __restrict__ W_rbf,
                 float*       __restrict__ hnode) {
    __shared__ float sW[HIDDEN * NUM_RAD];
    for (int t = threadIdx.x; t < HIDDEN * NUM_RAD; t += blockDim.x)
        sW[t] = W_rbf[t];
    __syncthreads();

    const float4* x4 = (const float4*)x;
    const int egrp = threadIdx.x >> 6;      // 4 edges per block-iteration
    const int cg   = threadIdx.x & 63;      // 64 channel-groups of 4
    const int c0   = cg * 4;

    // per-thread slice of W_rbf: 4 channels x 6 radial
    float w[4][NUM_RAD];
    #pragma unroll
    for (int j = 0; j < 4; ++j)
        #pragma unroll
        for (int r = 0; r < NUM_RAD; ++r)
            w[j][r] = sW[(c0 + j) * NUM_RAD + r];

    for (int e = blockIdx.x * 4 + egrp; e < E_EDGES; e += gridDim.x * 4) {
        float rb[NUM_RAD];
        #pragma unroll
        for (int r = 0; r < NUM_RAD; ++r) rb[r] = rbf[e * NUM_RAD + r];

        float4 xv = x4[(size_t)e * 64 + cg];

        float cf[4];
        #pragma unroll
        for (int j = 0; j < 4; ++j) {
            float s = 0.f;
            #pragma unroll
            for (int r = 0; r < NUM_RAD; ++r) s += w[j][r] * rb[r];
            cf[j] = s;
        }

        float* base = hnode + (size_t)idx[e] * HIDDEN + c0;
        unsafeAtomicAdd(base + 0, cf[0] * xv.x);
        unsafeAtomicAdd(base + 1, cf[1] * xv.y);
        unsafeAtomicAdd(base + 2, cf[2] * xv.z);
        unsafeAtomicAdd(base + 3, cf[3] * xv.w);
    }
}

// ---------------------------------------------------------------------------
// Kernel 3: fused node-side stack.  One block = 32 nodes x 256 channels.
// Activations ping-pong in LDS as bf16 (stride-padded), selected by integer
// offset (no LDS pointer arrays -- those break the static initializer).
// 8 waves: wave>>2 picks row half (16 rows), wave&3 picks 64-column slab
// (4 N-tiles of 16).  Each layer: D(16x16) += A(16x32 bf16) x B(32x16 bf16)
// over 8 K-steps using v_wmma_f32_16x16x32_bf16.  Epilogue adds bias
// (per-lane scalar since lane fixes the output column) and applies silu for
// layers 1..3.  Finale: 256->1 dot with W_out.
// ---------------------------------------------------------------------------
__global__ __launch_bounds__(256, 2)
void mlp_kernel(const float*  __restrict__ hnode,
                const __bf16* __restrict__ Bpack,
                const float*  __restrict__ b_up,
                const float*  __restrict__ bs,
                const float*  __restrict__ W_out,
                float*        __restrict__ out) {
    __shared__ __align__(16) __bf16 smem[2 * LDS_HALF];

    const int tid      = threadIdx.x;
    const int lane     = tid & 31;
    const int wave     = tid >> 5;
    const int rowBase  = (wave >> 2) * 16;   // 0 or 16
    const int colGroup = wave & 3;           // 0..3 -> 64-col slab
    const int gRow0    = blockIdx.x * 32;

    // stage input tile: f32 -> bf16 into LDS (buffer 0)
    #pragma unroll 4
    for (int it = 0; it < 32; ++it)
        smem[it * LDS_STRIDE + tid] =
            (__bf16)hnode[(size_t)(gRow0 + it) * HIDDEN + tid];
    __syncthreads();

    // A-fragment addressing (contiguous K order, same as Bpack): lane<16 -> row lane,
    // K 0..15; lane>=16 -> row lane-16, K 16..31.
    const int aRow  = rowBase + (lane & 15);
    const int aKoff = (lane < 16) ? 0 : 16;

    int cur = 0;
    for (int layer = 0; layer < 4; ++layer) {
        const int inOff  = cur ? LDS_HALF : 0;
        const int outOff = cur ? 0 : LDS_HALF;
        const float* bias = (layer == 0) ? b_up : (bs + (layer - 1) * HIDDEN);

        v8f acc[4] = {};
        for (int kk = 0; kk < 8; ++kk) {
            FragBF A;
            const __bf16* ap = &smem[inOff + aRow * LDS_STRIDE + kk * 32 + aKoff];
            A.u[0] = *(const v4u*)(ap);        // K 0..7   (ds_load_b128)
            A.u[1] = *(const v4u*)(ap + 8);    // K 8..15
            #pragma unroll
            for (int nt = 0; nt < 4; ++nt) {
                FragBF B;
                const __bf16* bp = Bpack
                    + (((size_t)(layer * 8 + kk) * 16 + (colGroup * 4 + nt)) * 512)
                    + lane * 16;               // 32 contiguous bytes per lane
                B.u[0] = *(const v4u*)(bp);
                B.u[1] = *(const v4u*)(bp + 8);
                acc[nt] = __builtin_amdgcn_wmma_f32_16x16x32_bf16(
                    false, A.v, false, B.v, (short)0, acc[nt], false, false);
            }
        }

        // epilogue: C/D layout -> lane fixes column, VGPR r fixes row
        #pragma unroll
        for (int nt = 0; nt < 4; ++nt) {
            const int col = colGroup * 64 + nt * 16 + (lane & 15);
            const float bv = bias[col];
            #pragma unroll
            for (int r = 0; r < 8; ++r) {
                float v = acc[nt][r] + bv;
                if (layer > 0) v = v * (1.0f / (1.0f + __expf(-v)));  // silu
                const int row = rowBase + ((lane < 16) ? r : r + 8);
                smem[outOff + row * LDS_STRIDE + col] = (__bf16)v;
            }
        }
        __syncthreads();
        cur ^= 1;
    }

    // final linear 256 -> 1 (no bias): 32 lanes, one node each.
    // After 4 layers cur==0, so result sits in buffer 0; use cur anyway.
    const int hfOff = cur ? LDS_HALF : 0;
    if (tid < 32) {
        const int g = gRow0 + tid;
        if (g < N_NODES) {
            float s = 0.f;
            #pragma unroll 8
            for (int c = 0; c < HIDDEN; ++c)
                s += (float)smem[hfOff + tid * LDS_STRIDE + c] * W_out[c];
            out[g] = s;
        }
    }
}

// ---------------------------------------------------------------------------
// Launch
// ---------------------------------------------------------------------------
extern "C" void kernel_launch(void* const* d_in, const int* in_sizes, int n_in,
                              void* d_out, int out_size, void* d_ws, size_t ws_size,
                              hipStream_t stream) {
    const float* x     = (const float*)d_in[0];
    const float* rbf   = (const float*)d_in[1];
    const int*   idx   = (const int*)  d_in[2];
    // d_in[3] = num_nodes scalar (compile-time constant here)
    const float* W_rbf = (const float*)d_in[4];
    const float* W_up  = (const float*)d_in[5];
    const float* b_up  = (const float*)d_in[6];
    const float* Ws    = (const float*)d_in[7];
    const float* bs    = (const float*)d_in[8];
    const float* W_out = (const float*)d_in[9];
    float* out = (float*)d_out;

    // workspace: [ h_nodes f32 N_PAD x 256 | Bpack bf16 4 x 256 x 256 ]
    float*  hnode = (float*)d_ws;
    __bf16* Bpack = (__bf16*)((char*)d_ws + (size_t)N_PAD * HIDDEN * sizeof(float));

    const int n4 = N_PAD * HIDDEN / 4;
    zero_kernel<<<(n4 + 255) / 256, 256, 0, stream>>>((float4*)hnode, n4);
    pack_weights_kernel<<<(4 * 65536 + 255) / 256, 256, 0, stream>>>(W_up, Ws, Bpack);
    edge_kernel<<<2048, 256, 0, stream>>>(x, rbf, idx, W_rbf, hnode);
    mlp_kernel<<<N_PAD / 32, 256, 0, stream>>>(hnode, Bpack, b_up, bs, W_out, out);
}
